// RPN_34024730919630
// MI455X (gfx1250) — matrix operand order repaired and checked by reading, not compiled, not run on hardware
//
#include <hip/hip_runtime.h>
#include <math.h>
#include <stdint.h>

typedef __bf16 bf16_t;
typedef bf16_t v16bf __attribute__((ext_vector_type(16)));
typedef bf16_t v8bf  __attribute__((ext_vector_type(8)));
typedef float  v8f   __attribute__((ext_vector_type(8)));
typedef unsigned int v4u __attribute__((ext_vector_type(4)));
typedef int v8i __attribute__((ext_vector_type(8)));
typedef int v4i __attribute__((ext_vector_type(4)));

#define A_TOTAL   159882
#define K_TOT     4507
#define K_PAD     8192
#define K_PAD2    4512
#define BBOX_CLIP 4.135166556742356f
#define LVL_OFF   1e5f
#define NMS_TH    0.7f

// d_out layout (float elements)
#define OUT_BOX   0
#define OUT_SCORE 8000
#define OUT_VALID 10000
#define OUT_PROP  12000
#define OUT_OBJ   1291056

__device__ __forceinline__ unsigned fkey(float f) {
  unsigned u = __float_as_uint(f);
  return (u & 0x80000000u) ? ~u : (u | 0x80000000u);
}

// ---------------- weight / activation conversion ----------------
__global__ void k_cvt_w(const float* __restrict__ w, bf16_t* __restrict__ wbf) {
  int t = blockIdx.x * blockDim.x + threadIdx.x;
  if (t >= 256 * 2304) return;
  int cout = t / 2304, k = t % 2304;
  int tap = k / 256, cin = k % 256;
  int cy = tap / 3, cx = tap % 3;
  wbf[(size_t)cout * 2304 + k] = (bf16_t)w[((cout * 256 + cin) * 3 + cy) * 3 + cx];
}

__global__ void k_cvt_hw(const float* __restrict__ cls_w, const float* __restrict__ reg_w,
                         bf16_t* __restrict__ hw) {
  int t = blockIdx.x * blockDim.x + threadIdx.x;
  if (t >= 16 * 256) return;
  int o = t / 256, c = t % 256;
  float v = 0.f;
  if (o < 3) v = cls_w[o * 256 + c];
  else if (o < 15) v = reg_w[(o - 3) * 256 + c];
  hw[t] = (bf16_t)v;
}

__global__ void k_cvt_x(const float* __restrict__ x, bf16_t* __restrict__ xbf,
                        int H, int W) {
  long t = (long)blockIdx.x * blockDim.x + threadIdx.x;
  long total = (long)2 * 256 * H * W;
  if (t >= total) return;
  int c = (int)(t & 255);
  long p = t >> 8;                      // (b*H + y)*W + x
  int xx = (int)(p % W);
  int yy = (int)((p / W) % H);
  int b  = (int)(p / ((long)H * W));
  xbf[(size_t)p * 256 + c] = (bf16_t)x[(((size_t)(b * 256 + c) * H) + yy) * W + xx];
}

// ---------------- 3x3 conv + ReLU via bf16 WMMA (implicit GEMM) ----------------
// Per block: 256 couts x 32 positions. Per tap, the 256x256 weight slice for that
// tap is DMA'd into LDS by the Tensor Data Mover; A-fragments then come from LDS.
__global__ void __launch_bounds__(128)
k_conv3x3(const bf16_t* __restrict__ xbf, const bf16_t* __restrict__ wbf,
          const float* __restrict__ bias, bf16_t* __restrict__ tbf,
          int H, int W, int P) {
  __shared__ bf16_t wtap[256 * 256];   // 128 KB staged weight slice (one 3x3 tap)
  const int tid   = threadIdx.x;
  const int lane  = tid & 31;
  const int wave  = tid >> 5;
  const int n     = lane & 15;       // N column (position within 16-wide tile)
  const int khalf = lane >> 4;       // which K-half this lane holds
  const int cout_w = wave * 64;
  const int HW = H * W;

  long p0 = (long)blockIdx.x * 32;
  long pp[2];
  int bb[2], ybase[2], xbase[2];
#pragma unroll
  for (int t = 0; t < 2; t++) {
    long p = p0 + t * 16 + n;
    pp[t] = p;
    long pc = (p < P) ? p : (P - 1);   // clamp: EXEC must stay all-ones for WMMA
    int b = (int)(pc / HW);
    int rr = (int)(pc % HW);
    bb[t] = b; ybase[t] = rr / W; xbase[t] = rr % W;
  }

  v8f acc[2][4];
#pragma unroll
  for (int t = 0; t < 2; t++)
#pragma unroll
    for (int s = 0; s < 4; s++)
#pragma unroll
      for (int r = 0; r < 8; r++) acc[t][s][r] = 0.0f;

#pragma unroll 1
  for (int tap = 0; tap < 9; ++tap) {
    __syncthreads();                 // previous tap's LDS reads complete
    if (wave == 0) {
      // ---- Tensor Data Mover: 2D tile [256 rows x 256 elems], 2B elements,
      //      row stride 2304 elems, from wbf + tap*256 into LDS offset 0.
      unsigned long long ga = (unsigned long long)(uintptr_t)(wbf + (size_t)tap * 256);
      v4u g0;
      g0[0] = 1u;                                        // count=1, user descriptor
      g0[1] = 0u;                                        // lds_addr = 0 (wtap)
      g0[2] = (unsigned)(ga & 0xFFFFFFFFu);              // global_addr[31:0]
      g0[3] = (unsigned)((ga >> 32) & 0x1FFFFFFu) | (2u << 30);  // addr[56:32] | type=2
      v8i g1;
      g1[0] = (int)(1u << 16);                           // data_size = 1 (2 bytes)
      g1[1] = (int)(2304u << 16);                        // tensor_dim0[15:0] in [31:16]
      g1[2] = (int)(256u << 16);                         // tensor_dim0 hi=0 | tensor_dim1 lo=256
      g1[3] = (int)(256u << 16);                         // tensor_dim1 hi=0 | tile_dim0=256
      g1[4] = (int)(256u);                               // tile_dim1=256 | tile_dim2=0
      g1[5] = (int)(2304);                               // tensor_dim0_stride[31:0]
      g1[6] = 0;                                         // stride hi | dim1_stride lo
      g1[7] = 0;
      v4i gz4 = {0, 0, 0, 0};                            // groups 2/3: unused (<=2D)
      v8i gz8 = {0, 0, 0, 0, 0, 0, 0, 0};                // extra group (clang-23 form)
      __builtin_amdgcn_tensor_load_to_lds(g0, g1, gz4, gz4, gz8, 0);
      __builtin_amdgcn_s_wait_tensorcnt(0);
    }
    __syncthreads();                 // staged weights visible to all waves

    const int dy = tap / 3 - 1, dx = tap % 3 - 1;
    const bf16_t* xrow[2];
    bool inb[2];
#pragma unroll
    for (int t = 0; t < 2; t++) {
      int yy = ybase[t] + dy, xx = xbase[t] + dx;
      inb[t] = (yy >= 0) && (yy < H) && (xx >= 0) && (xx < W);
      xrow[t] = xbf + (size_t)((bb[t] * H + yy) * (long)W + xx) * 256;
    }
    // prefetch next tap's activation row for tile 0
    __builtin_prefetch(xrow[0] + 256, 0, 0);

#pragma unroll 1
    for (int kb = 0; kb < 8; ++kb) {
      // ---- B fragments: 32x16, lane holds K = khalf*16 .. +15 for column n
      v16bf bfr[2];
#pragma unroll
      for (int t = 0; t < 2; t++) {
        if (inb[t]) {
          const bf16_t* src = xrow[t] + kb * 32 + khalf * 16;
          v8bf l0 = *(const v8bf*)(src);
          v8bf l1 = *(const v8bf*)(src + 8);
#pragma unroll
          for (int i = 0; i < 8; i++) { bfr[t][i] = l0[i]; bfr[t][i + 8] = l1[i]; }
        } else {
#pragma unroll
          for (int i = 0; i < 16; i++) bfr[t][i] = (bf16_t)0.0f;
        }
      }
      // ---- 4 A fragments (64 couts) from LDS, each feeding 2 WMMAs
#pragma unroll
      for (int s = 0; s < 4; s++) {
        const int m = cout_w + s * 16 + n;               // A row = cout
        const bf16_t* lw = wtap + (size_t)m * 256 + kb * 32 + khalf * 16;
        v8bf a0 = *(const v8bf*)(lw);
        v8bf a1 = *(const v8bf*)(lw + 8);
        v16bf af;
#pragma unroll
        for (int i = 0; i < 8; i++) { af[i] = a0[i]; af[i + 8] = a1[i]; }
        acc[0][s] = __builtin_amdgcn_wmma_f32_16x16x32_bf16(
            false, af, false, bfr[0], (short)0, acc[0][s], false, false);
        acc[1][s] = __builtin_amdgcn_wmma_f32_16x16x32_bf16(
            false, af, false, bfr[1], (short)0, acc[1][s], false, false);
      }
    }
  }

#pragma unroll
  for (int t = 0; t < 2; t++) {
    if (pp[t] < P) {
#pragma unroll
      for (int s = 0; s < 4; s++) {
        const int cbase = cout_w + s * 16 + khalf * 8;   // D: VGPR r -> M = r + 8*khalf
        v8bf outv;
#pragma unroll
        for (int r = 0; r < 8; r++) {
          float v = acc[t][s][r] + bias[cbase + r];
          outv[r] = (bf16_t)(v > 0.0f ? v : 0.0f);
        }
        *(v8bf*)(tbf + (size_t)pp[t] * 256 + cbase) = outv;
      }
    }
  }
}

// ---------------- fused 1x1 heads + anchor decode ----------------
__global__ void k_heads(const bf16_t* __restrict__ tbf, const bf16_t* __restrict__ hw,
                        const float* __restrict__ cls_b, const float* __restrict__ reg_b,
                        float* __restrict__ out, int H, int W, int P,
                        float stride, float asize, int anch_off) {
  int p = blockIdx.x * blockDim.x + threadIdx.x;
  if (p >= P) return;
  float acc[15];
#pragma unroll
  for (int o = 0; o < 3; o++)  acc[o] = cls_b[o];
#pragma unroll
  for (int o = 0; o < 12; o++) acc[3 + o] = reg_b[o];
  const bf16_t* trow = tbf + (size_t)p * 256;
  for (int cb = 0; cb < 32; ++cb) {
    v8bf t8 = *(const v8bf*)(trow + cb * 8);
    float f[8];
#pragma unroll
    for (int i = 0; i < 8; i++) f[i] = (float)t8[i];
#pragma unroll
    for (int o = 0; o < 15; o++) {
      v8bf w8 = *(const v8bf*)(hw + o * 256 + cb * 8);
      float s = 0.f;
#pragma unroll
      for (int i = 0; i < 8; i++) s += f[i] * (float)w8[i];
      acc[o] += s;
    }
  }
  int HW = H * W;
  int b = p / HW, rr = p % HW, y = rr / W, x = rr % W;
  const float ratios[3] = {0.5f, 1.0f, 2.0f};
  float cxa = (x + 0.5f) * stride, cya = (y + 0.5f) * stride;
#pragma unroll
  for (int a = 0; a < 3; a++) {
    float sq = sqrtf(ratios[a]);
    float wa = asize * sq, ha = asize / sq;
    float dxv = acc[3 + 4 * a], dyv = acc[4 + 4 * a];
    float dwv = fminf(acc[5 + 4 * a], BBOX_CLIP);
    float dhv = fminf(acc[6 + 4 * a], BBOX_CLIP);
    float cx = dxv * wa + cxa, cy = dyv * ha + cya;
    float pwd = expf(dwv) * wa, phd = expf(dhv) * ha;
    int g = anch_off + (y * W + x) * 3 + a;
    float* pr = out + OUT_PROP + ((size_t)b * A_TOTAL + g) * 4;
    pr[0] = cx - 0.5f * pwd; pr[1] = cy - 0.5f * phd;
    pr[2] = cx + 0.5f * pwd; pr[3] = cy + 0.5f * phd;
    out[OUT_OBJ + (size_t)b * A_TOTAL + g] = acc[a];
  }
}

// ---------------- per-(image,level) deterministic top-k ----------------
__global__ void k_zero(int* cnt) { if (threadIdx.x < 16) cnt[threadIdx.x] = 0; }

__global__ void k_topk(const float* __restrict__ out, const float* __restrict__ shapes,
                       float* __restrict__ cscore, int* __restrict__ cgidx,
                       int* __restrict__ clvl, float4* __restrict__ cbox,
                       int* __restrict__ cnt) {
  const int NLn[5]  = {120000, 30000, 7500, 1875, 507};
  const int AOFF[5] = {0, 120000, 150000, 157500, 159375};
  const int KL[5]   = {1000, 1000, 1000, 1000, 507};
  const int KOFF[5] = {0, 1000, 2000, 3000, 4000};
  int b = blockIdx.x / 5, lvl = blockIdx.x % 5;
  int n = NLn[lvl], k = KL[lvl];
  const float* sc = out + OUT_OBJ + (size_t)b * A_TOTAL + AOFF[lvl];
  __shared__ unsigned scnt;
  unsigned long long prefix = 0;
  for (int bit = 63; bit >= 0; --bit) {
    unsigned long long cand = prefix | (1ull << bit);
    if (threadIdx.x == 0) scnt = 0;
    __syncthreads();
    unsigned local = 0;
    for (int i = threadIdx.x; i < n; i += blockDim.x) {
      unsigned long long key =
          ((unsigned long long)fkey(sc[i]) << 32) | (0xFFFFFFFFull - (unsigned)i);
      if (key >= cand) local++;
    }
    atomicAdd(&scnt, local);
    __syncthreads();
    if (scnt >= (unsigned)k) prefix = cand;
    __syncthreads();
  }
  // exactly k keys satisfy key >= prefix (keys are distinct by construction)
  float imgw = shapes[b * 2 + 0], imgh = shapes[b * 2 + 1];
  for (int i = threadIdx.x; i < n; i += blockDim.x) {
    unsigned long long key =
        ((unsigned long long)fkey(sc[i]) << 32) | (0xFFFFFFFFull - (unsigned)i);
    if (key >= prefix) {
      int slot = atomicAdd(&cnt[b * 5 + lvl], 1);
      int idx = KOFF[lvl] + slot;
      int g = AOFF[lvl] + i;
      const float* pr = out + OUT_PROP + ((size_t)b * A_TOTAL + g) * 4;
      float x1 = fminf(fmaxf(pr[0], 0.f), imgw);
      float y1 = fminf(fmaxf(pr[1], 0.f), imgh);
      float x2 = fminf(fmaxf(pr[2], 0.f), imgw);
      float y2 = fminf(fmaxf(pr[3], 0.f), imgh);
      bool valid = ((x2 - x1) > 1.0f) && ((y2 - y1) > 1.0f);
      size_t base = (size_t)b * K_PAD;
      cscore[base + idx] = valid ? sc[i] : -INFINITY;
      cgidx[base + idx] = g;
      clvl[base + idx] = lvl;
      cbox[base + idx] = make_float4(x1, y1, x2, y2);
    }
  }
}

// ---------------- per-image bitonic sort (score desc, gidx asc) ----------------
__global__ void k_sort(const float* __restrict__ cscore, const int* __restrict__ cgidx,
                       const int* __restrict__ clvl, const float4* __restrict__ cbox,
                       float4* __restrict__ sbox, int* __restrict__ slvl,
                       float* __restrict__ sscore) {
  __shared__ unsigned long long sk[K_PAD];  // 64 KB LDS
  int b = blockIdx.x;
  size_t cb = (size_t)b * K_PAD;
  for (int i = threadIdx.x; i < K_PAD; i += blockDim.x) {
    unsigned long long key = 0;
    if (i < K_TOT) {
      float s = cscore[cb + i];
      int g = cgidx[cb + i];
      key = ((unsigned long long)fkey(s) << 32) |
            ((unsigned long long)(unsigned)(262143 - g) << 14) |
            (unsigned long long)(unsigned)i;   // slot in low 14 bits
    }
    sk[i] = key;
  }
  __syncthreads();
  for (unsigned ksz = 2; ksz <= K_PAD; ksz <<= 1) {
    for (unsigned j = ksz >> 1; j > 0; j >>= 1) {
      for (unsigned idx = threadIdx.x; idx < K_PAD; idx += blockDim.x) {
        unsigned ixj = idx ^ j;
        if (ixj > idx) {
          bool desc = ((idx & ksz) == 0);
          unsigned long long a = sk[idx], c = sk[ixj];
          bool sw = desc ? (a < c) : (a > c);
          if (sw) { sk[idx] = c; sk[ixj] = a; }
        }
      }
      __syncthreads();
    }
  }
  size_t sb = (size_t)b * K_PAD2;
  for (int i = threadIdx.x; i < K_TOT; i += blockDim.x) {
    int slot = (int)(sk[i] & 0x3FFFull);
    sbox[sb + i] = cbox[cb + slot];
    slvl[sb + i] = clvl[cb + slot];
    sscore[sb + i] = cscore[cb + slot];
  }
}

// ---------------- per-image NMS (level-offset trick) ----------------
__global__ void k_nms(const float4* __restrict__ sbox, const int* __restrict__ slvl,
                      const float* __restrict__ sscore, int* __restrict__ keepg) {
  __shared__ int keep[K_TOT];
  int b = blockIdx.x;
  size_t sb = (size_t)b * K_PAD2;
  for (int i = threadIdx.x; i < K_TOT; i += blockDim.x)
    keep[i] = (sscore[sb + i] > -INFINITY) ? 1 : 0;
  __syncthreads();
  for (int i = 0; i < K_TOT; i++) {
    if (keep[i]) {
      float4 bi = sbox[sb + i];
      float oi = (float)slvl[sb + i] * LVL_OFF;
      float ix1 = bi.x + oi, iy1 = bi.y + oi, ix2 = bi.z + oi, iy2 = bi.w + oi;
      float ai = (ix2 - ix1) * (iy2 - iy1);
      for (int j = i + 1 + threadIdx.x; j < K_TOT; j += blockDim.x) {
        if (keep[j]) {
          float4 bj = sbox[sb + j];
          float oj = (float)slvl[sb + j] * LVL_OFF;
          float jx1 = bj.x + oj, jy1 = bj.y + oj, jx2 = bj.z + oj, jy2 = bj.w + oj;
          float aj = (jx2 - jx1) * (jy2 - jy1);
          float lx = fmaxf(ix1, jx1), ly = fmaxf(iy1, jy1);
          float rx = fminf(ix2, jx2), ry = fminf(iy2, jy2);
          float iw = fmaxf(rx - lx, 0.f), ih = fmaxf(ry - ly, 0.f);
          float inter = iw * ih;
          float iou = inter / (ai + aj - inter + 1e-9f);
          if (iou > NMS_TH) keep[j] = 0;
        }
      }
    }
    __syncthreads();
  }
  for (int i = threadIdx.x; i < K_TOT; i += blockDim.x)
    keepg[b * K_PAD2 + i] = keep[i];
}

// ---------------- finalize: replicate top_k over (kept..., -inf...) ----------------
__global__ void k_final(const float4* __restrict__ sbox, const float* __restrict__ sscore,
                        const int* __restrict__ keepg, float* __restrict__ out) {
  int b = blockIdx.x;
  if (threadIdx.x != 0) return;
  size_t sb = (size_t)b * K_PAD2;
  const int* kp = keepg + b * K_PAD2;
  int outp = 0;
  for (int pass = 0; pass < 2 && outp < 1000; ++pass) {
    int want = (pass == 0) ? 1 : 0;
    for (int i = 0; i < K_TOT && outp < 1000; ++i) {
      if (kp[i] == want) {
        float4 bb = sbox[sb + i];
        float* bo = out + OUT_BOX + ((size_t)b * 1000 + outp) * 4;
        bo[0] = bb.x; bo[1] = bb.y; bo[2] = bb.z; bo[3] = bb.w;
        out[OUT_SCORE + b * 1000 + outp] = want ? sscore[sb + i] : -INFINITY;
        out[OUT_VALID + b * 1000 + outp] = want ? 1.0f : 0.0f;
        outp++;
      }
    }
  }
}

// ---------------- host-side launch ----------------
extern "C" void kernel_launch(void* const* d_in, const int* in_sizes, int n_in,
                              void* d_out, int out_size, void* d_ws, size_t ws_size,
                              hipStream_t stream) {
  (void)in_sizes; (void)n_in; (void)out_size; (void)ws_size;
  const float* xs[5] = {(const float*)d_in[0], (const float*)d_in[1],
                        (const float*)d_in[2], (const float*)d_in[3],
                        (const float*)d_in[4]};
  const float* conv_w = (const float*)d_in[5];
  const float* conv_b = (const float*)d_in[6];
  const float* cls_w  = (const float*)d_in[7];
  const float* cls_b  = (const float*)d_in[8];
  const float* reg_w  = (const float*)d_in[9];
  const float* reg_b  = (const float*)d_in[10];
  const float* shapes = (const float*)d_in[11];
  float* out = (float*)d_out;
  char* ws = (char*)d_ws;

  bf16_t* wbf   = (bf16_t*)(ws + 0);            // 256*2304 bf16
  bf16_t* hwbf  = (bf16_t*)(ws + 1179648);      // 16*256 bf16
  bf16_t* xbf   = (bf16_t*)(ws + 1187840);      // level activations (NHWC bf16)
  bf16_t* tbf   = (bf16_t*)(ws + 42147840);     // conv output (NHWC bf16)
  float*  cscore = (float*)(ws + 83107840);     // [2][8192]
  int*    cgidx  = (int*)  (ws + 83173376);
  int*    clvl   = (int*)  (ws + 83238912);
  float4* cbox   = (float4*)(ws + 83304448);    // [2][8192]
  int*    cnt    = (int*)  (ws + 83566592);     // 10 counters
  float4* sbox   = (float4*)(ws + 83566848);    // [2][4512]
  int*    slvl   = (int*)  (ws + 83711232);
  float*  sscore = (float*)(ws + 83747328);
  int*    keepg  = (int*)  (ws + 83783424);

  k_zero<<<1, 32, 0, stream>>>(cnt);
  k_cvt_w<<<(256 * 2304 + 255) / 256, 256, 0, stream>>>(conv_w, wbf);
  k_cvt_hw<<<16, 256, 0, stream>>>(cls_w, reg_w, hwbf);

  static const int   HL[5] = {200, 100, 50, 25, 13};
  static const float ST[5] = {4.f, 8.f, 16.f, 32.f, 64.f};
  static const float SZ[5] = {32.f, 64.f, 128.f, 256.f, 512.f};
  static const int   AO[5] = {0, 120000, 150000, 157500, 159375};

  for (int l = 0; l < 5; ++l) {
    int H = HL[l], W = HL[l];
    int P = 2 * H * W;
    long elems = (long)P * 256;
    k_cvt_x<<<(unsigned)((elems + 255) / 256), 256, 0, stream>>>(xs[l], xbf, H, W);
    k_conv3x3<<<(P + 31) / 32, 128, 0, stream>>>(xbf, wbf, conv_b, tbf, H, W, P);
    k_heads<<<(P + 255) / 256, 256, 0, stream>>>(tbf, hwbf, cls_b, reg_b, out,
                                                 H, W, P, ST[l], SZ[l], AO[l]);
  }

  k_topk<<<10, 256, 0, stream>>>(out, shapes, cscore, cgidx, clvl, cbox, cnt);
  k_sort<<<2, 256, 0, stream>>>(cscore, cgidx, clvl, cbox, sbox, slvl, sscore);
  k_nms<<<2, 256, 0, stream>>>(sbox, slvl, sscore, keepg);
  k_final<<<2, 32, 0, stream>>>(sbox, sscore, keepg, out);
}
